// PermutationClosedStructureInverse_37486474560134
// MI455X (gfx1250) — matrix operand order repaired and checked by reading, not compiled, not run on hardware
//
#include <hip/hip_runtime.h>

#define B_     1024
#define R_     128
#define NPCS   4
#define NCONST 2
#define CIN    64
#define COUT   64
#define KDIM   256      // NPCS*CIN
#define XROWS  514      // NPCS*R_ + NCONST
#define RSX    264      // padded K-stride (bf16 elems) for X rows: 528B -> 4-dword bank skew
#define RSW    264      // padded K-stride for W^T rows

typedef __attribute__((ext_vector_type(16))) __bf16 v16bf;
typedef __attribute__((ext_vector_type(8)))  __bf16 v8bf;
typedef __attribute__((ext_vector_type(8)))  float  v8f;
typedef __attribute__((ext_vector_type(4)))  float  v4f;

__device__ __forceinline__ void split_bf16(float v, __bf16& h, __bf16& l) {
    h = (__bf16)v;                 // RNE high part
    l = (__bf16)(v - (float)h);    // residual
}

__global__ __launch_bounds__(1024, 1)
void pcs_fused_kernel(const float* __restrict__ x,
                      const float* __restrict__ W,
                      const float* __restrict__ V,
                      float* __restrict__ out) {
    __shared__ __align__(16) __bf16 sXhi[R_ * RSX];    // X[n][K] hi   (67.6 KB)
    __shared__ __align__(16) __bf16 sXlo[R_ * RSX];    // X[n][K] lo   (67.6 KB)
    __shared__ __align__(16) __bf16 sWhi[COUT * RSW];  // Wd^T[o][K] hi (33.8 KB)
    __shared__ __align__(16) __bf16 sWlo[COUT * RSW];  // Wd^T[o][K] lo (33.8 KB)
    __shared__ float sT[KDIM];
    __shared__ float sBias[COUT];

    const int b   = blockIdx.x;
    const int tid = threadIdx.x;
    const float* __restrict__ xb = x + (size_t)b * (XROWS * CIN);

    // ---- Phase 1: stream x[b] rows 0..511 -> LDS as bf16 hi/lo, X[n][K=g*64+c]
    #pragma unroll
    for (int j = 0; j < 8; ++j) {
        int idx = (j * 1024 + tid) * 4;            // float index in [0, 512*64)
        v4f v = *(const v4f*)(xb + idx);
        int r = idx >> 6;                          // row in [0,512)
        int c = idx & 63;
        int n = r & (R_ - 1);
        int K = ((r >> 7) << 6) + c;
        int o = n * RSX + K;
        #pragma unroll
        for (int e = 0; e < 4; ++e) {
            __bf16 h, l;
            split_bf16(v[e], h, l);
            sXhi[o + e] = h;
            sXlo[o + e] = l;
        }
    }

    // ---- Phase 2: Wd = W[:,0]-W[:,1], store transposed [o][K] as bf16 hi/lo
    #pragma unroll
    for (int j = 0; j < 16; ++j) {
        int i = j * 1024 + tid;                    // i in [0, 4*64*64)
        int g = i >> 12;
        int o = (i >> 6) & 63;
        int c = i & 63;
        float w0 = W[(g * 2 + 0) * (COUT * CIN) + o * CIN + c];
        float w1 = W[(g * 2 + 1) * (COUT * CIN) + o * CIN + c];
        __bf16 h, l;
        split_bf16(w0 - w1, h, l);
        int K = (g << 6) + c;
        sWhi[o * RSW + K] = h;
        sWlo[o * RSW + K] = l;
    }
    __syncthreads();

    // ---- Phase 3: t[K] = sum_n X[n][K]  (reconstruct fp32 as hi+lo)
    if (tid < KDIM) {
        float acc = 0.f;
        for (int n = 0; n < R_; ++n)
            acc += (float)sXhi[n * RSX + tid] + (float)sXlo[n * RSX + tid];
        sT[tid] = acc;
    }
    __syncthreads();

    // ---- Phase 4: bias[o] = sum_K t[K]*W1[K,o] + sum_{k,c} xc[b,k,c]*V[k,o,c]
    if (tid < COUT) {
        const int o = tid;
        float acc = 0.f;
        for (int K = 0; K < KDIM; ++K) {
            int g = K >> 6, c = K & 63;
            acc += sT[K] * W[(g * 2 + 1) * (COUT * CIN) + o * CIN + c];
        }
        for (int k = 0; k < NCONST; ++k)
            for (int c = 0; c < CIN; ++c)
                acc += xb[(NPCS * R_ + k) * CIN + c] * V[k * (COUT * CIN) + o * CIN + c];
        sBias[o] = acc;
    }
    __syncthreads();

    // ---- Phase 5: 32 waves, one 16x16 output tile each; bf16x3 WMMA over K=256
    const int lane = tid & 31;
    const int wave = tid >> 5;
    const int m0   = (wave >> 2) * 16;     // 8 M-tiles
    const int o0   = (wave & 3) * 16;      // 4 N-tiles
    const int l15  = lane & 15;
    const int hi16 = (lane >> 4) & 1;

    v8f acc;
    {
        float bias = sBias[o0 + l15];      // C/D column = o0 + (lane&15) for all 8 regs
        #pragma unroll
        for (int v = 0; v < 8; ++v) acc[v] = bias;
    }

    const int arow = m0 + l15;             // A: row = m0 + (lane&15)
    const int bcol = o0 + l15;             // B: col = o0 + (lane&15)

    #pragma unroll
    for (int ks = 0; ks < 8; ++ks) {
        const int k0 = ks * 32;
        const int kA = k0 + hi16 * 8;      // A lanes>=16 start at K+8
        const int kB = k0 + hi16 * 8;      // B lanes>=16 start at K+8

        v8bf ah0 = *(const v8bf*)&sXhi[arow * RSX + kA];
        v8bf ah1 = *(const v8bf*)&sXhi[arow * RSX + kA + 16];
        v8bf al0 = *(const v8bf*)&sXlo[arow * RSX + kA];
        v8bf al1 = *(const v8bf*)&sXlo[arow * RSX + kA + 16];
        v8bf bh0 = *(const v8bf*)&sWhi[bcol * RSW + kB];
        v8bf bh1 = *(const v8bf*)&sWhi[bcol * RSW + kB + 16];
        v8bf bl0 = *(const v8bf*)&sWlo[bcol * RSW + kB];
        v8bf bl1 = *(const v8bf*)&sWlo[bcol * RSW + kB + 16];

        v16bf ahi = __builtin_shufflevector(ah0, ah1, 0,1,2,3,4,5,6,7,8,9,10,11,12,13,14,15);
        v16bf alo = __builtin_shufflevector(al0, al1, 0,1,2,3,4,5,6,7,8,9,10,11,12,13,14,15);
        v16bf bhi = __builtin_shufflevector(bh0, bh1, 0,1,2,3,4,5,6,7,8,9,10,11,12,13,14,15);
        v16bf blo = __builtin_shufflevector(bl0, bl1, 0,1,2,3,4,5,6,7,8,9,10,11,12,13,14,15);

        acc = __builtin_amdgcn_wmma_f32_16x16x32_bf16(false, ahi, false, bhi, (short)0, acc, false, false);
        acc = __builtin_amdgcn_wmma_f32_16x16x32_bf16(false, ahi, false, blo, (short)0, acc, false, false);
        acc = __builtin_amdgcn_wmma_f32_16x16x32_bf16(false, alo, false, bhi, (short)0, acc, false, false);
    }

    // ---- Store: D vgpr v -> row m0 + hi16*8 + v, col o0 + (lane&15)
    float* __restrict__ ob = out + (size_t)b * (R_ * COUT);
    const int orow = m0 + hi16 * 8;
    #pragma unroll
    for (int v = 0; v < 8; ++v)
        ob[(orow + v) * COUT + o0 + l15] = acc[v];
}

extern "C" void kernel_launch(void* const* d_in, const int* in_sizes, int n_in,
                              void* d_out, int out_size, void* d_ws, size_t ws_size,
                              hipStream_t stream) {
    (void)in_sizes; (void)n_in; (void)d_ws; (void)ws_size; (void)out_size;
    const float* x = (const float*)d_in[0];
    const float* W = (const float*)d_in[1];
    const float* V = (const float*)d_in[2];
    float* out = (float*)d_out;
    pcs_fused_kernel<<<dim3(B_), dim3(1024), 0, stream>>>(x, W, V, out);
}
